// SpanRankingSRLDecoder_11751030522853
// MI455X (gfx1250) — matrix operand order repaired and checked by reading, not compiled, not run on hardware
//
#include <hip/hip_runtime.h>
#include <hip/hip_bf16.h>

// ---------------------------------------------------------------------------
// Problem constants
// ---------------------------------------------------------------------------
constexpr int B_    = 8;
constexpr int L_    = 128;
constexpr int D_    = 512;
constexpr int MAW_  = 8;
constexpr int SWF_  = 20;
constexpr int FF_   = 512;
constexpr int R_    = 19;
constexpr int G_    = 4;
constexpr int SPAN_  = 3 * D_ + SWF_;   // 1556
constexpr int SPAN1_ = SPAN_ + 1;       // 1557
constexpr int SPANP_ = 1600;            // K padded to 64 (25*64)
constexpr int S_     = L_ * MAW_;       // 1024 spans / batch
constexpr int BS_    = B_ * S_;         // 8192
constexpr int BGN_   = G_ * R_;         // 76 biaffine cols
constexpr int BGNP_  = 128;             // padded to 64
constexpr int WPNP_  = 64;              // srl_wp cols padded to 64

typedef __bf16 bf16;
typedef __attribute__((ext_vector_type(16))) __bf16 v16bf;
typedef __attribute__((ext_vector_type(8)))  __bf16 v8bf;
typedef __attribute__((ext_vector_type(8)))  float  v8f;

union AFrag { v16bf v; v8bf h[2]; };

__device__ __host__ inline bf16 f2bf(float f) {
  union { float f; unsigned u; } x; x.f = f;
  unsigned r = x.u + 0x7fffu + ((x.u >> 16) & 1u);
  unsigned short hs = (unsigned short)(r >> 16);
  bf16 out;
  __builtin_memcpy(&out, &hs, 2);
  return out;
}

// ---------------------------------------------------------------------------
// Pack fp32 weight (K x N, opt transposed source) into bf16 WMMA-B fragment
// layout. Tile (kt,nt) of 32x16 holds 512 bf16: lane = (n&15) + 16*((k>>3)&1),
// elem = (k&7) + 8*(k>=16) (ISA 7.12.2 16-bit B layout). Zero-pads k>=srcK, n>=srcN.
// Kp % 64 == 0, Npad % 64 == 0 required by the GEMM.
// ---------------------------------------------------------------------------
__global__ void pack_b_kernel(const float* __restrict__ src, bf16* __restrict__ dst,
                              int srcK, int srcN, int Kp, int Npad, int transpose,
                              long srcBatchStride, long dstBatchStride) {
  const int batch = blockIdx.z;
  src += (long)batch * srcBatchStride;
  dst += (long)batch * dstBatchStride;
  long idx = (long)blockIdx.x * blockDim.x + threadIdx.x;
  long total = (long)Kp * Npad;
  if (idx >= total) return;
  int k = (int)(idx / Npad);
  int n = (int)(idx % Npad);
  float v = 0.f;
  if (k < srcK && n < srcN)
    v = transpose ? src[(long)n * srcK + k] : src[(long)k * srcN + n];
  int kt = k >> 5, klocal = k & 31;
  int nt = n >> 4;
  int lane = (n & 15) + (((klocal >> 3) & 1) << 4);
  int e = (klocal & 7) + ((klocal >> 4) << 3);
  long off = ((long)kt * (Npad >> 4) + nt) * 512 + lane * 16 + e;
  dst[off] = f2bf(v);
}

// ---------------------------------------------------------------------------
// Build span1 rows (bf16, K padded to 1600):
//   [ start_emb(512) | end_emb(512) | width_emb(20) | head_emb(512) | 1 | 0.. ]
// One block per span; 8 candidate tokens staged in LDS.
// ---------------------------------------------------------------------------
__global__ void __launch_bounds__(256)
span_kernel(const float* __restrict__ hs, const int* __restrict__ slen,
            const float* __restrict__ wtab, bf16* __restrict__ spanA) {
  const int bs = blockIdx.x;
  const int b = bs >> 10;          // / S_
  const int s = bs & (S_ - 1);
  const int pos  = s & (L_ - 1);   // s = wIdx*L + pos
  const int wIdx = s >> 7;
  const int cand = (pos + wIdx) < slen[b];
  const int st = cand ? pos : 0;
  const int w  = cand ? wIdx : 0;

  __shared__ float tok[MAW_ * D_];
  for (int i = threadIdx.x; i < MAW_ * D_; i += blockDim.x) {
    int j = i >> 9;                // / D_
    int d = i & (D_ - 1);
    int rowi = st + j; if (rowi > L_ - 1) rowi = L_ - 1;
    tok[i] = hs[((long)b * L_ + rowi) * D_ + d];
  }
  __syncthreads();

  bf16* outp = spanA + (long)bs * SPANP_;
  for (int c = threadIdx.x; c < SPANP_; c += blockDim.x) {
    float v;
    if (c < D_) {
      v = tok[c];                                   // start emb = tok row 0
    } else if (c < 2 * D_) {
      v = tok[w * D_ + (c - D_)];                   // end emb = tok row w
    } else if (c < 2 * D_ + SWF_) {
      v = wtab[w * SWF_ + (c - 2 * D_)];            // width emb
    } else if (c < SPAN_) {
      int d = c - (2 * D_ + SWF_);
      float sum = 0.f;
      for (int j = 0; j <= w; ++j) sum += tok[j * D_ + d];
      v = sum * (1.f / MAW_);                       // head emb (mean over MAW)
    } else if (c == SPAN_) {
      v = 1.f;                                      // ones column (span1)
    } else {
      v = 0.f;                                      // K padding
    }
    outp[c] = f2bf(v);
  }
}

// ---------------------------------------------------------------------------
// WMMA bf16 GEMM: C[M,Npad] = act(A[M,Kp] @ Bpacked + bias)
// 128 threads = 4 waves; wave -> 32x32 tile; block tile 64x64; batched (grid.z)
// Requirements: M % 64 == 0, Kp % 64 == 0, Npad % 64 == 0 (no guards needed).
// Software-pipelined: next-iteration fragments load while current WMMAs run.
// ---------------------------------------------------------------------------
__device__ inline v8f wmma_bf16(v16bf a, v16bf b, v8f c) {
  return __builtin_amdgcn_wmma_f32_16x16x32_bf16(false, a, false, b, (short)0, c,
                                                 false, false);
}

__device__ inline void store_tile(void* C, const float* bias, v8f c,
                                  int m0, int n0, int ldc,
                                  int lane, int do_relu, int out_f32) {
  int col = n0 + (lane & 15);
  int row0 = m0 + ((lane >> 4) << 3);
  float bv = bias ? bias[col] : 0.f;
#pragma unroll
  for (int i = 0; i < 8; ++i) {
    int row = row0 + i;
    float v = c[i] + bv;
    if (do_relu && v < 0.f) v = 0.f;
    if (out_f32) ((float*)C)[(long)row * ldc + col] = v;
    else         ((bf16*)C)[(long)row * ldc + col] = f2bf(v);
  }
}

__global__ void __launch_bounds__(128)
gemm_wmma_bf16(const bf16* __restrict__ A, const bf16* __restrict__ Bp,
               const float* __restrict__ bias, void* __restrict__ C,
               int M, int Kp, int Npad, int ldc, int do_relu, int out_f32,
               long sA, long sB, long sC) {
  const int batch = blockIdx.z;
  A  += (long)batch * sA;
  Bp += (long)batch * sB;
  void* Cb = out_f32 ? (void*)((float*)C + (long)batch * sC)
                     : (void*)((bf16*)C + (long)batch * sC);

  const int lane = threadIdx.x & 31;
  const int wave = threadIdx.x >> 5;
  const int m0 = blockIdx.y * 64 + (wave >> 1) * 32;
  const int n0 = blockIdx.x * 64 + (wave & 1) * 32;

  const int ntiles = Npad >> 4;
  const long bstep = (long)ntiles * 512;   // packed elems per 32-row K step

  const int r  = lane & 15;
  const int kg = lane >> 4;

  v8f c00 = {}, c01 = {}, c10 = {}, c11 = {};

  const bf16* aptr = A + (long)(m0 + r) * Kp + kg * 8;
  const long arow16 = (long)16 * Kp;
  const bf16* bptr = Bp + (long)(n0 >> 4) * 512 + lane * 16;

  // prologue: load k0 = 0 fragments
  AFrag a0c, a1c;
  a0c.h[0] = *(const v8bf*)(aptr);
  a0c.h[1] = *(const v8bf*)(aptr + 16);
  a1c.h[0] = *(const v8bf*)(aptr + arow16);
  a1c.h[1] = *(const v8bf*)(aptr + arow16 + 16);
  v16bf b0c = *(const v16bf*)(bptr);
  v16bf b1c = *(const v16bf*)(bptr + 512);

  for (int k0 = 32; k0 < Kp; k0 += 32) {
    aptr += 32;
    bptr += bstep;
    AFrag a0n, a1n;
    a0n.h[0] = *(const v8bf*)(aptr);
    a0n.h[1] = *(const v8bf*)(aptr + 16);
    a1n.h[0] = *(const v8bf*)(aptr + arow16);
    a1n.h[1] = *(const v8bf*)(aptr + arow16 + 16);
    v16bf b0n = *(const v16bf*)(bptr);
    v16bf b1n = *(const v16bf*)(bptr + 512);
    __builtin_prefetch(aptr + 256, 0, 1);   // global_prefetch_b8 on A stream

    c00 = wmma_bf16(a0c.v, b0c, c00);
    c01 = wmma_bf16(a0c.v, b1c, c01);
    c10 = wmma_bf16(a1c.v, b0c, c10);
    c11 = wmma_bf16(a1c.v, b1c, c11);

    a0c = a0n; a1c = a1n; b0c = b0n; b1c = b1n;
  }
  c00 = wmma_bf16(a0c.v, b0c, c00);
  c01 = wmma_bf16(a0c.v, b1c, c01);
  c10 = wmma_bf16(a1c.v, b0c, c10);
  c11 = wmma_bf16(a1c.v, b1c, c11);

  store_tile(Cb, bias, c00, m0,      n0,      ldc, lane, do_relu, out_f32);
  store_tile(Cb, bias, c01, m0,      n0 + 16, ldc, lane, do_relu, out_f32);
  store_tile(Cb, bias, c10, m0 + 16, n0,      ldc, lane, do_relu, out_f32);
  store_tile(Cb, bias, c11, m0 + 16, n0 + 16, ldc, lane, do_relu, out_f32);
}

// ---------------------------------------------------------------------------
// Predicate-side small work (32 rows): pred FFN score, p = pred_emb@srl_w0[SPAN:],
// pred1 = [relu(pred_emb@scale_w + scale_b), 1, 0-pad]
// ---------------------------------------------------------------------------
__global__ void __launch_bounds__(256)
pred_kernel(const float* __restrict__ hs, const int* __restrict__ gold,
            const float* __restrict__ w0, const float* __restrict__ b0,
            const float* __restrict__ w1, const float* __restrict__ b1,
            const float* __restrict__ wp, const float* __restrict__ bp,
            const float* __restrict__ srl_w0, const float* __restrict__ scale_w,
            const float* __restrict__ scale_b,
            float* __restrict__ pred_scores, float* __restrict__ pbuf,
            float* __restrict__ pred1) {
  const int bg = blockIdx.x;
  const int b = bg / G_;
  const int tid = threadIdx.x;

  __shared__ float pe[D_], h0[FF_], h1[FF_], red[256];

  const int gp = gold[bg];
  for (int d = tid; d < D_; d += 256) pe[d] = hs[((long)b * L_ + gp) * D_ + d];
  __syncthreads();

  for (int f = tid; f < FF_; f += 256) {
    float s = b0[f];
    for (int d = 0; d < D_; ++d) s += pe[d] * w0[(long)d * FF_ + f];
    h0[f] = s > 0.f ? s : 0.f;
  }
  __syncthreads();
  for (int f = tid; f < FF_; f += 256) {
    float s = b1[f];
    for (int d = 0; d < FF_; ++d) s += h0[d] * w1[(long)d * FF_ + f];
    h1[f] = s > 0.f ? s : 0.f;
  }
  __syncthreads();
  float s = 0.f;
  for (int f = tid; f < FF_; f += 256) s += h1[f] * wp[f];
  red[tid] = s;
  __syncthreads();
  for (int st = 128; st > 0; st >>= 1) {
    if (tid < st) red[tid] += red[tid + st];
    __syncthreads();
  }
  if (tid == 0) pred_scores[bg] = red[0] + bp[0];

  for (int f = tid; f < FF_; f += 256) {
    float s2 = 0.f;
    for (int d = 0; d < D_; ++d) s2 += pe[d] * srl_w0[(long)(SPAN_ + d) * FF_ + f];
    pbuf[(long)bg * FF_ + f] = s2;
  }
  for (int c = tid; c < SPANP_; c += 256) {
    float v;
    if (c < SPAN_) {
      float s3 = scale_b[c];
      for (int d = 0; d < D_; ++d) s3 += pe[d] * scale_w[(long)d * SPAN_ + c];
      v = s3 > 0.f ? s3 : 0.f;
    } else if (c == SPAN_) v = 1.f;
    else v = 0.f;
    pred1[(long)bg * SPANP_ + c] = v;
  }
}

// ---------------------------------------------------------------------------
// t[bg, r*SPAN1+i] = sum_j pred1[bg,j] * biaffine_w[r,i,j]   (~3 GFLOP)
// ---------------------------------------------------------------------------
__global__ void __launch_bounds__(256)
t_kernel(const float* __restrict__ pred1, const float* __restrict__ bw,
         float* __restrict__ tbuf) {
  const int bg = blockIdx.y;
  __shared__ float p1[SPAN1_];
  for (int i = threadIdx.x; i < SPAN1_; i += 256)
    p1[i] = pred1[(long)bg * SPANP_ + i];
  __syncthreads();
  long ri = (long)blockIdx.x * 256 + threadIdx.x;
  if (ri >= (long)R_ * SPAN1_) return;
  const float* wrow = bw + ri * SPAN1_;   // (r*SPAN1+i) row, contiguous in j
  float s = 0.f;
  for (int j = 0; j < SPAN1_; ++j) s += p1[j] * wrow[j];
  tbuf[(long)bg * (R_ * SPAN1_) + ri] = s;
}

// ---------------------------------------------------------------------------
// h = relu(a[bs,f] + p[b,g,f] + srl_b0[f])  -> bf16 (32768 x 512)
// ---------------------------------------------------------------------------
__global__ void hbuild_kernel(const float* __restrict__ a, const float* __restrict__ p,
                              const float* __restrict__ b0, bf16* __restrict__ h) {
  long idx = (long)blockIdx.x * blockDim.x + threadIdx.x;
  if (idx >= (long)BS_ * G_ * FF_) return;
  int f = (int)(idx & (FF_ - 1));
  int g = (int)((idx >> 9) & (G_ - 1));
  long bs = idx >> 11;                 // / (FF_*G_)
  int b = (int)(bs >> 10);
  float v = a[bs * FF_ + f] + p[((long)b * G_ + g) * FF_ + f] + b0[f];
  h[idx] = f2bf(v > 0.f ? v : 0.f);
}

// ---------------------------------------------------------------------------
// Row dot: out[row] = sum_k A[row,k]*w[k] + bias  (wave32 shuffle reduce)
// ---------------------------------------------------------------------------
__global__ void __launch_bounds__(256)
rowdot_kernel(const bf16* __restrict__ A, const float* __restrict__ w,
              const float* __restrict__ bias, float* __restrict__ out,
              int M, int K) {
  const int wave = threadIdx.x >> 5;
  const int lane = threadIdx.x & 31;
  const int row = blockIdx.x * 8 + wave;
  if (row >= M) return;
  float s = 0.f;
  for (int k = lane; k < K; k += 32) s += (float)A[(long)row * K + k] * w[k];
  for (int m = 16; m > 0; m >>= 1) s += __shfl_xor(s, m, 32);
  if (lane == 0) out[row] = s + bias[0];
}

// ---------------------------------------------------------------------------
// Final combine + mask
// ---------------------------------------------------------------------------
__global__ void combine_kernel(const float* __restrict__ srluC,
                               const float* __restrict__ biaffC,
                               const float* __restrict__ argsc,
                               const float* __restrict__ predsc,
                               const float* __restrict__ srl_bp,
                               const int* __restrict__ slen,
                               float* __restrict__ out) {
  long idx = (long)blockIdx.x * blockDim.x + threadIdx.x;
  if (idx >= (long)B_ * S_ * G_ * R_) return;
  int r = (int)(idx % R_);
  long t1 = idx / R_;
  int g = (int)(t1 % G_);
  long t2 = t1 / G_;
  int s = (int)(t2 % S_);
  int b = (int)(t2 / S_);
  int pos = s & (L_ - 1), wIdx = s >> 7;
  float v = srluC[(((long)(b * S_ + s) * G_ + g) << 6) + r] + srl_bp[r]
          + biaffC[(((long)b * S_ + s) << 7) + g * R_ + r]
          + argsc[b * S_ + s] + predsc[b * G_ + g];
  out[idx] = ((pos + wIdx) < slen[b]) ? v : 0.f;
}

// ---------------------------------------------------------------------------
// Launcher
// ---------------------------------------------------------------------------
extern "C" void kernel_launch(void* const* d_in, const int* in_sizes, int n_in,
                              void* d_out, int out_size, void* d_ws, size_t ws_size,
                              hipStream_t stream) {
  (void)in_sizes; (void)n_in; (void)out_size; (void)ws_size;

  const float* hs      = (const float*)d_in[0];
  const int*   slen    = (const int*)  d_in[1];
  const int*   gold    = (const int*)  d_in[2];
  const float* wtab    = (const float*)d_in[3];
  const float* arg_w0  = (const float*)d_in[4];
  const float* arg_b0  = (const float*)d_in[5];
  const float* arg_w1  = (const float*)d_in[6];
  const float* arg_b1  = (const float*)d_in[7];
  const float* arg_wp  = (const float*)d_in[8];
  const float* arg_bp  = (const float*)d_in[9];
  const float* pred_w0 = (const float*)d_in[10];
  const float* pred_b0 = (const float*)d_in[11];
  const float* pred_w1 = (const float*)d_in[12];
  const float* pred_b1 = (const float*)d_in[13];
  const float* pred_wp = (const float*)d_in[14];
  const float* pred_bp = (const float*)d_in[15];
  const float* srl_w0  = (const float*)d_in[16];
  const float* srl_b0  = (const float*)d_in[17];
  const float* srl_w1  = (const float*)d_in[18];
  const float* srl_b1  = (const float*)d_in[19];
  const float* srl_wp  = (const float*)d_in[20];
  const float* srl_bp  = (const float*)d_in[21];
  const float* scale_w = (const float*)d_in[22];
  const float* scale_b = (const float*)d_in[23];
  const float* biaf_w  = (const float*)d_in[24];
  float* out = (float*)d_out;

  // workspace carve-up
  char* base = (char*)d_ws;
  size_t off = 0;
  auto take = [&](size_t bytes) -> void* {
    off = (off + 255) & ~(size_t)255;
    void* p = base + off;
    off += bytes;
    return p;
  };
  const long tpackB = (long)(SPANP_ / 32) * (BGNP_ / 16) * 512;  // per batch elems

  bf16*  spanA   = (bf16*) take((size_t)BS_ * SPANP_ * 2);
  bf16*  argw0p  = (bf16*) take((size_t)SPANP_ * FF_ * 2);
  bf16*  srlw0p  = (bf16*) take((size_t)SPANP_ * FF_ * 2);
  bf16*  argw1p  = (bf16*) take((size_t)FF_ * FF_ * 2);
  bf16*  srlw1p  = (bf16*) take((size_t)FF_ * FF_ * 2);
  bf16*  srlwpp  = (bf16*) take((size_t)FF_ * WPNP_ * 2);
  bf16*  argh0   = (bf16*) take((size_t)BS_ * FF_ * 2);
  bf16*  argh1   = (bf16*) take((size_t)BS_ * FF_ * 2);
  float* abuf    = (float*)take((size_t)BS_ * FF_ * 4);
  float* pbuf    = (float*)take((size_t)B_ * G_ * FF_ * 4);
  float* pred1   = (float*)take((size_t)B_ * G_ * SPANP_ * 4);
  float* predsc  = (float*)take((size_t)B_ * G_ * 4);
  float* argsc   = (float*)take((size_t)BS_ * 4);
  float* tbuf    = (float*)take((size_t)B_ * G_ * R_ * SPAN1_ * 4);
  bf16*  tpack   = (bf16*) take((size_t)B_ * tpackB * 2);
  bf16*  hbuf    = (bf16*) take((size_t)BS_ * G_ * FF_ * 2);
  bf16*  h2buf   = (bf16*) take((size_t)BS_ * G_ * FF_ * 2);
  float* srluC   = (float*)take((size_t)BS_ * G_ * WPNP_ * 4);
  float* biaffC  = (float*)take((size_t)B_ * S_ * BGNP_ * 4);

  auto packGrid = [](long total) { return dim3((unsigned)((total + 255) / 256), 1, 1); };

  // ---- weight packing (fp32 -> bf16 fragment layout, N padded to x64) ----
  pack_b_kernel<<<packGrid((long)SPANP_ * FF_), 256, 0, stream>>>(
      arg_w0, argw0p, SPAN_, FF_, SPANP_, FF_, 0, 0, 0);
  pack_b_kernel<<<packGrid((long)SPANP_ * FF_), 256, 0, stream>>>(
      srl_w0, srlw0p, SPAN_, FF_, SPANP_, FF_, 0, 0, 0);
  pack_b_kernel<<<packGrid((long)FF_ * FF_), 256, 0, stream>>>(
      arg_w1, argw1p, FF_, FF_, FF_, FF_, 0, 0, 0);
  pack_b_kernel<<<packGrid((long)FF_ * FF_), 256, 0, stream>>>(
      srl_w1, srlw1p, FF_, FF_, FF_, FF_, 0, 0, 0);
  pack_b_kernel<<<packGrid((long)FF_ * WPNP_), 256, 0, stream>>>(
      srl_wp, srlwpp, FF_, R_, FF_, WPNP_, 0, 0, 0);

  // ---- span embeddings (span1, bf16, K-padded) ----
  span_kernel<<<BS_, 256, 0, stream>>>(hs, slen, wtab, spanA);

  // ---- predicate-side small work ----
  pred_kernel<<<B_ * G_, 256, 0, stream>>>(hs, gold, pred_w0, pred_b0, pred_w1,
                                           pred_b1, pred_wp, pred_bp, srl_w0,
                                           scale_w, scale_b, predsc, pbuf, pred1);

  // ---- biaffine t, then pack per-batch B operand (K=i, N=g*R+r) ----
  t_kernel<<<dim3((unsigned)((R_ * SPAN1_ + 255) / 256), B_ * G_, 1), 256, 0, stream>>>(
      pred1, biaf_w, tbuf);
  {
    dim3 g((unsigned)(((long)SPANP_ * BGNP_ + 255) / 256), 1, B_);
    pack_b_kernel<<<g, 256, 0, stream>>>(tbuf, tpack, SPAN1_, BGN_, SPANP_, BGNP_, 1,
                                         (long)G_ * R_ * SPAN1_, tpackB);
  }

  // ---- WMMA GEMMs ----
  // arg branch: h0 = relu(span@arg_w0+b0); h1 = relu(h0@arg_w1+b1)
  gemm_wmma_bf16<<<dim3(FF_ / 64, BS_ / 64, 1), 128, 0, stream>>>(
      spanA, argw0p, arg_b0, argh0, BS_, SPANP_, FF_, FF_, 1, 0, 0, 0, 0);
  gemm_wmma_bf16<<<dim3(FF_ / 64, BS_ / 64, 1), 128, 0, stream>>>(
      argh0, argw1p, arg_b1, argh1, BS_, FF_, FF_, FF_, 1, 0, 0, 0, 0);
  rowdot_kernel<<<BS_ / 8, 256, 0, stream>>>(argh1, arg_wp, arg_bp, argsc, BS_, FF_);

  // srl branch: a = span@srl_w0[:SPAN] (f32, no act)
  gemm_wmma_bf16<<<dim3(FF_ / 64, BS_ / 64, 1), 128, 0, stream>>>(
      spanA, srlw0p, nullptr, abuf, BS_, SPANP_, FF_, FF_, 0, 1, 0, 0, 0);
  // h = relu(a + p + b0) over (b,s,g)
  hbuild_kernel<<<(unsigned)(((long)BS_ * G_ * FF_ + 255) / 256), 256, 0, stream>>>(
      abuf, pbuf, srl_b0, hbuf);
  // h2 = relu(h@srl_w1 + b1)
  gemm_wmma_bf16<<<dim3(FF_ / 64, (BS_ * G_) / 64, 1), 128, 0, stream>>>(
      hbuf, srlw1p, srl_b1, h2buf, BS_ * G_, FF_, FF_, FF_, 1, 0, 0, 0, 0);
  // srl_unary (pre-bias) = h2@srl_wp  (f32, N padded to 64)
  gemm_wmma_bf16<<<dim3(WPNP_ / 64, (BS_ * G_) / 64, 1), 128, 0, stream>>>(
      h2buf, srlwpp, nullptr, srluC, BS_ * G_, FF_, WPNP_, WPNP_, 0, 1, 0, 0, 0);

  // biaffine: per-batch span1 @ t  (M=1024, K=1600, N=128 padded)
  gemm_wmma_bf16<<<dim3(BGNP_ / 64, S_ / 64, B_), 128, 0, stream>>>(
      spanA, tpack, nullptr, biaffC, S_, SPANP_, BGNP_, BGNP_, 0, 1,
      (long)S_ * SPANP_, tpackB, (long)S_ * BGNP_);

  // ---- combine + mask ----
  combine_kernel<<<(unsigned)(((long)B_ * S_ * G_ * R_ + 255) / 256), 256, 0, stream>>>(
      srluC, biaffC, argsc, predsc, srl_bp, slen, out);
}
